// RGCNDecoder_10101763080456
// MI455X (gfx1250) — matrix operand order repaired and checked by reading, not compiled, not run on hardware
//
#include <hip/hip_runtime.h>
#include <hip/hip_bf16.h>

typedef _Float16 half_t;
typedef __attribute__((ext_vector_type(16))) _Float16 v16h;
typedef __attribute__((ext_vector_type(8)))  _Float16 v8h;
typedef __attribute__((ext_vector_type(8)))  float    v8f;

#define ALPHA 0.01f
#define BN_EPS 1e-5f

// ---------------------------------------------------------------------------
// Elementwise / setup kernels
// ---------------------------------------------------------------------------
__global__ void zero_f32_kernel(float* __restrict__ p, size_t n) {
    size_t i = (size_t)blockIdx.x * blockDim.x + threadIdx.x;
    size_t stride = (size_t)gridDim.x * blockDim.x;
    for (; i < n; i += stride) p[i] = 0.0f;
}

// A[n, k] for k in [0, (R+1)*Cin): zero relation blocks, copy x into root slot.
__global__ void init_A_kernel(float* __restrict__ A, const float* __restrict__ x,
                              int Nn, int Cin, int Rr) {
    const int Ktot = (Rr + 1) * Cin;
    size_t total = (size_t)Nn * Ktot;
    size_t i = (size_t)blockIdx.x * blockDim.x + threadIdx.x;
    size_t stride = (size_t)gridDim.x * blockDim.x;
    for (; i < total; i += stride) {
        size_t n = i / Ktot;
        int k = (int)(i - n * Ktot);
        A[i] = (k >= Rr * Cin) ? x[n * Cin + (k - Rr * Cin)] : 0.0f;
    }
}

__global__ void deg_count_kernel(const int* __restrict__ tgt, const int* __restrict__ et,
                                 float* __restrict__ norm, int E, int Rr) {
    int i = blockIdx.x * blockDim.x + threadIdx.x;
    if (i < E) atomicAdd(&norm[(size_t)tgt[i] * Rr + et[i]], 1.0f);
}

__global__ void deg_invert_kernel(float* __restrict__ norm, size_t n) {
    size_t i = (size_t)blockIdx.x * blockDim.x + threadIdx.x;
    size_t stride = (size_t)gridDim.x * blockDim.x;
    for (; i < n; i += stride) {
        float d = norm[i];
        norm[i] = 1.0f / fmaxf(d, 1.0f);
    }
}

// One block per edge, one thread per input channel: A[tgt, rel*Cin + c] += w * x[src, c]
__global__ void aggregate_kernel(const int* __restrict__ src, const int* __restrict__ tgt,
                                 const int* __restrict__ et, const float* __restrict__ norm,
                                 const float* __restrict__ x, float* __restrict__ A,
                                 int Cin, int Rr) {
    int e = blockIdx.x;
    int c = threadIdx.x;
    int s = src[e], t = tgt[e], r = et[e];
    float w = norm[(size_t)t * Rr + r];
    int Ktot = (Rr + 1) * Cin;
    atomicAdd(&A[(size_t)t * Ktot + (size_t)r * Cin + c], w * x[(size_t)s * Cin + c]);
}

// Stacked RGCN weight, transposed + f16: Wt[o][k], k = r*Cin+i for r<R, root block after.
__global__ void build_w_kernel(const float* __restrict__ basis, const float* __restrict__ comp,
                               const float* __restrict__ root, half_t* __restrict__ Wt,
                               int Cin, int Cout, int Bb, int Rr) {
    const int Ktot = (Rr + 1) * Cin;
    size_t total = (size_t)Ktot * Cout;
    size_t i = (size_t)blockIdx.x * blockDim.x + threadIdx.x;
    size_t stride = (size_t)gridDim.x * blockDim.x;
    for (; i < total; i += stride) {
        int o = (int)(i % Cout);
        int k = (int)(i / Cout);
        float v;
        if (k >= Rr * Cin) {
            v = root[(size_t)(k - Rr * Cin) * Cout + o];
        } else {
            int r = k / Cin, ii = k - r * Cin;
            v = 0.0f;
            for (int b = 0; b < Bb; ++b)
                v += comp[r * Bb + b] * basis[((size_t)b * Cin + ii) * Cout + o];
        }
        Wt[(size_t)o * Ktot + k] = (half_t)v;
    }
}

// Wt[o][k] = (f16) W[k][o]
__global__ void transpose_to_half_kernel(const float* __restrict__ W, half_t* __restrict__ Wt,
                                         int K, int Cout) {
    size_t total = (size_t)K * Cout;
    size_t i = (size_t)blockIdx.x * blockDim.x + threadIdx.x;
    size_t stride = (size_t)gridDim.x * blockDim.x;
    for (; i < total; i += stride) {
        int o = (int)(i % Cout);
        int k = (int)(i / Cout);
        Wt[(size_t)o * K + k] = (half_t)W[(size_t)k * Cout + o];
    }
}

// ---------------------------------------------------------------------------
// WMMA GEMM: out[M,Cout] = f16(A[M,K]) @ Wt^T + bias.
//   A : f32 row-major [M,K]  (converted to f16 on the fly)
//   Wt: f16 [Cout][K]        (B operand, column-of-output major)
// Block = 128 threads = 4 waves. Wave -> 16 rows x 128 cols (8 WMMA tiles).
// Block -> 64 rows x 128 cols. B tile (32K x 128 cols) double-buffered in LDS
// (one barrier per K-step; next tile's global loads in flight across the
// WMMA burst). K % 32 == 0, Cout % 128 == 0, K/32 >= 2.
// ---------------------------------------------------------------------------
#define BS_STRIDE 40          // halfs per column: 80B, 16B-aligned, conflict-free
#define BS_BUF    (128 * BS_STRIDE)

__global__ __launch_bounds__(128)
void wmma_gemm_kernel(const float* __restrict__ A, const half_t* __restrict__ Wt,
                      const float* __restrict__ bias, float* __restrict__ out,
                      int M, int K, int Cout) {
    __shared__ __align__(16) half_t Bs[2 * BS_BUF];

    const int lane   = threadIdx.x & 31;
    const int wave   = threadIdx.x >> 5;
    const int hi     = lane >> 4;       // 0: lanes 0-15, 1: lanes 16-31
    const int l16    = lane & 15;
    const int rowTile = blockIdx.x * 64 + wave * 16;
    const int colBase = blockIdx.y * 128;
    const int aRow    = rowTile + l16;               // logical A row (M = l16)
    const int aRowC   = (aRow < M) ? aRow : (M - 1); // clamped: loads always valid
    const int kb      = hi * 8;  // ISA f16 A layout: lo lanes K 0..7,16..23; hi 8..15,24..31

    // cooperative-fill mapping: 512 chunks of 8 halfs; 4 chunks per column
    const int fillCol0 = threadIdx.x >> 2;           // 0..31; +g*32 covers 128 cols
    const int fillKc   = (threadIdx.x & 3) * 8;

    v8f acc[8];
#pragma unroll
    for (int s = 0; s < 8; ++s)
#pragma unroll
        for (int j = 0; j < 8; ++j) acc[s][j] = 0.f;

    const float* aBase = A + (size_t)aRowC * K;
    const int nsteps = K >> 5;

    // ---- prefill buffer 0 with K-step 0 ----
#pragma unroll
    for (int g = 0; g < 4; ++g) {
        int col = fillCol0 + g * 32;
        const half_t* gp = Wt + (size_t)(colBase + col) * K + fillKc;
        *(v8h*)&Bs[col * BS_STRIDE + fillKc] = *(const v8h*)gp;
    }
    __syncthreads();

    for (int step = 0; step < nsteps; ++step) {
        const int k0  = step << 5;
        const int cur = (step & 1) * BS_BUF;
        const int nxt = BS_BUF - cur;
        const int k0n = (step + 1 < nsteps) ? (k0 + 32) : k0;  // clamped (last store unused)

        // ---- (a) issue next B tile global loads (in flight during WMMAs) ----
        v8h r0, r1, r2, r3;
        {
            const size_t base = (size_t)colBase * K + k0n + fillKc;
            r0 = *(const v8h*)(Wt + base + (size_t)(fillCol0 +  0) * K);
            r1 = *(const v8h*)(Wt + base + (size_t)(fillCol0 + 32) * K);
            r2 = *(const v8h*)(Wt + base + (size_t)(fillCol0 + 64) * K);
            r3 = *(const v8h*)(Wt + base + (size_t)(fillCol0 + 96) * K);
        }

        // ---- (b) A fragment (f32 -> f16 on the fly, no divergence) ----
        const float* ap = aBase + k0 + kb;
        __builtin_prefetch(ap + 32, 0, 0);           // next K-step of A (speculative)
        v16h a;
#pragma unroll
        for (int j = 0; j < 8; ++j) {
            a[j]     = (_Float16)ap[j];              // K = k0+kb+0..7
            a[8 + j] = (_Float16)ap[16 + j];         // K = k0+kb+16..23
        }

        // ---- (c) 8 column subtiles from current LDS buffer ----
#pragma unroll
        for (int sub = 0; sub < 8; ++sub) {
            const half_t* bp = &Bs[cur + (sub * 16 + l16) * BS_STRIDE + hi * 8];
            v8h blo = *(const v8h*)bp;
            v8h bhi = *(const v8h*)(bp + 16);
            v16h b;
#pragma unroll
            for (int j = 0; j < 8; ++j) { b[j] = blo[j]; b[8 + j] = bhi[j]; }
            acc[sub] = __builtin_amdgcn_wmma_f32_16x16x32_f16(
                false, a, false, b, (short)0, acc[sub], false, false);
        }

        // ---- (d) commit next tile into other buffer, one barrier per step ----
        *(v8h*)&Bs[nxt + (fillCol0 +  0) * BS_STRIDE + fillKc] = r0;
        *(v8h*)&Bs[nxt + (fillCol0 + 32) * BS_STRIDE + fillKc] = r1;
        *(v8h*)&Bs[nxt + (fillCol0 + 64) * BS_STRIDE + fillKc] = r2;
        *(v8h*)&Bs[nxt + (fillCol0 + 96) * BS_STRIDE + fillKc] = r3;
        __syncthreads();
    }

    // ---- Epilogue: C/D layout: VGPR v, lane l -> M = v + 8*(l>=16), N = l%16 ----
#pragma unroll
    for (int sub = 0; sub < 8; ++sub) {
        int col = colBase + sub * 16 + l16;
        float bv = bias[col];
#pragma unroll
        for (int v = 0; v < 8; ++v) {
            int row = rowTile + v + hi * 8;
            if (row < M) out[(size_t)row * Cout + col] = acc[sub][v] + bv;
        }
    }
}

// ---------------------------------------------------------------------------
// BatchNorm (training stats) + LeakyReLU
// ---------------------------------------------------------------------------
__global__ void bn_stats_kernel(const float* __restrict__ H, float* __restrict__ sums,
                                int Nn, int Cout, int rowsPerBlock) {
    int o = threadIdx.x;
    int r0 = blockIdx.x * rowsPerBlock;
    int r1 = r0 + rowsPerBlock; if (r1 > Nn) r1 = Nn;
    float s = 0.f, s2 = 0.f;
    for (int r = r0; r < r1; ++r) {
        float v = H[(size_t)r * Cout + o];
        s += v; s2 += v * v;
    }
    atomicAdd(&sums[o], s);
    atomicAdd(&sums[Cout + o], s2);
}

__global__ void bn_apply_lrelu_kernel(const float* __restrict__ H, const float* __restrict__ sums,
                                      const float* __restrict__ gamma, const float* __restrict__ beta,
                                      float* __restrict__ X, int Nn, int Cout) {
    size_t total = (size_t)Nn * Cout;
    size_t i = (size_t)blockIdx.x * blockDim.x + threadIdx.x;
    size_t stride = (size_t)gridDim.x * blockDim.x;
    float invN = 1.0f / (float)Nn;
    for (; i < total; i += stride) {
        int o = (int)(i % Cout);
        float mu  = sums[o] * invN;
        float var = sums[Cout + o] * invN - mu * mu;
        float y = (H[i] - mu) * rsqrtf(var + BN_EPS) * gamma[o] + beta[o];
        X[i] = (y > 0.0f) ? y : ALPHA * y;
    }
}

// ---------------------------------------------------------------------------
// Host-side launcher
// ---------------------------------------------------------------------------
extern "C" void kernel_launch(void* const* d_in, const int* in_sizes, int n_in,
                              void* d_out, int out_size, void* d_ws, size_t ws_size,
                              hipStream_t stream) {
    const int R = 8, B = 30;
    const int N = in_sizes[0] / 128;   // 50000
    const int E = in_sizes[2];         // 800000

    const float* emb    = (const float*)d_in[0];
    const int*   eidx   = (const int*)d_in[1];
    const int*   etype  = (const int*)d_in[2];
    const float* basis1 = (const float*)d_in[3];
    const float* comp1  = (const float*)d_in[4];
    const float* root1  = (const float*)d_in[5];
    const float* bias1  = (const float*)d_in[6];
    const float* gamma1 = (const float*)d_in[7];
    const float* beta1  = (const float*)d_in[8];
    const float* basis2 = (const float*)d_in[9];
    const float* comp2  = (const float*)d_in[10];
    const float* root2  = (const float*)d_in[11];
    const float* bias2  = (const float*)d_in[12];
    const float* gamma2 = (const float*)d_in[13];
    const float* beta2  = (const float*)d_in[14];
    const float* Wf     = (const float*)d_in[15];
    const float* bf     = (const float*)d_in[16];

    const int* src = eidx;
    const int* tgt = eidx + E;

    // Workspace partition
    char* wsb = (char*)d_ws;
    size_t off = 0;
    auto carve = [&](size_t bytes) -> char* {
        char* p = wsb + off;
        off += (bytes + 255) & ~(size_t)255;
        return p;
    };
    float*  A    = (float*) carve((size_t)N * 2304 * sizeof(float));  // stacked agg, reused
    float*  Xb   = (float*) carve((size_t)N * 512 * sizeof(float));   // activations
    float*  Hb   = (float*) carve((size_t)N * 512 * sizeof(float));   // pre-BN GEMM out
    float*  norm = (float*) carve((size_t)N * R * sizeof(float));
    half_t* Wt   = (half_t*)carve((size_t)2304 * 512 * sizeof(half_t));
    float*  sums = (float*) carve(1024 * sizeof(float));
    (void)ws_size;

    const int TB = 256;
    const int GSB = 2048; // grid-stride blocks

    // ---- degrees / mean norm (shared by both layers) ----
    zero_f32_kernel<<<GSB, TB, 0, stream>>>(norm, (size_t)N * R);
    deg_count_kernel<<<(E + TB - 1) / TB, TB, 0, stream>>>(tgt, etype, norm, E, R);
    deg_invert_kernel<<<GSB, TB, 0, stream>>>(norm, (size_t)N * R);

    // =================== Layer 1: Cin=128 -> Cout=256, K=1152 ===================
    {
        const int Cin = 128, Cout = 256, K = (R + 1) * Cin;
        build_w_kernel<<<GSB, TB, 0, stream>>>(basis1, comp1, root1, Wt, Cin, Cout, B, R);
        init_A_kernel<<<GSB, TB, 0, stream>>>(A, emb, N, Cin, R);
        aggregate_kernel<<<E, Cin, 0, stream>>>(src, tgt, etype, norm, emb, A, Cin, R);
        dim3 grid((N + 63) / 64, Cout / 128);
        wmma_gemm_kernel<<<grid, 128, 0, stream>>>(A, Wt, bias1, Hb, N, K, Cout);
        zero_f32_kernel<<<8, TB, 0, stream>>>(sums, 2 * Cout);
        int statBlocks = 128, rowsPer = (N + statBlocks - 1) / statBlocks;
        bn_stats_kernel<<<statBlocks, Cout, 0, stream>>>(Hb, sums, N, Cout, rowsPer);
        bn_apply_lrelu_kernel<<<GSB, TB, 0, stream>>>(Hb, sums, gamma1, beta1, Xb, N, Cout);
    }

    // =================== Layer 2: Cin=256 -> Cout=512, K=2304 ===================
    {
        const int Cin = 256, Cout = 512, K = (R + 1) * Cin;
        build_w_kernel<<<GSB, TB, 0, stream>>>(basis2, comp2, root2, Wt, Cin, Cout, B, R);
        init_A_kernel<<<GSB, TB, 0, stream>>>(A, Xb, N, Cin, R);
        aggregate_kernel<<<E, Cin, 0, stream>>>(src, tgt, etype, norm, Xb, A, Cin, R);
        dim3 grid((N + 63) / 64, Cout / 128);
        wmma_gemm_kernel<<<grid, 128, 0, stream>>>(A, Wt, bias2, Hb, N, K, Cout);
        zero_f32_kernel<<<8, TB, 0, stream>>>(sums, 2 * Cout);
        int statBlocks = 128, rowsPer = (N + statBlocks - 1) / statBlocks;
        bn_stats_kernel<<<statBlocks, Cout, 0, stream>>>(Hb, sums, N, Cout, rowsPer);
        // x3 overwrites x2 in Xb (x2 no longer needed)
        bn_apply_lrelu_kernel<<<GSB, TB, 0, stream>>>(Hb, sums, gamma2, beta2, Xb, N, Cout);
    }

    // =================== Final linear: [N,512] @ [512,512] + bf ===================
    {
        const int K = 512, Cout = 512;
        transpose_to_half_kernel<<<GSB, TB, 0, stream>>>(Wf, Wt, K, Cout);
        dim3 grid((N + 63) / 64, Cout / 128);
        wmma_gemm_kernel<<<grid, 128, 0, stream>>>(Xb, Wt, bf, (float*)d_out, N, K, Cout);
    }
}
